// HierarchicalDecoder_20787641712757
// MI455X (gfx1250) — matrix operand order repaired and checked by reading, not compiled, not run on hardware
//
#include <hip/hip_runtime.h>
#include <hip/hip_bf16.h>
#include <math.h>

// ---------------------------------------------------------------------------
// Hierarchical LSTM decoder for MI455X (gfx1250, wave32, WMMA bf16).
// All GEMMs run as v_wmma_f32_16x16x32_bf16; weights converted once to bf16
// in workspace (~57MB, L2-resident on the 192MB L2); f32 accumulation;
// cell states kept f32. Compute-bound => matrix pipe is the right target.
// Wave tile 32x64 (8 accumulators) for ~21 FLOP/B operand reuse.
// ---------------------------------------------------------------------------

typedef __bf16 bf16;
typedef __attribute__((ext_vector_type(8)))  __bf16 v8bf;
typedef __attribute__((ext_vector_type(16))) __bf16 v16bf;
typedef __attribute__((ext_vector_type(8)))  float  v8f;

#define BATCH    1024
#define LATENT   512
#define CONDH    1024
#define CONDOUT  512
#define BOTH     1024
#define OUTD     90
#define NUM_SUB  16
#define SUB_LEN  4
#define SEQ      (NUM_SUB * SUB_LEN)   // 64
#define XDIM     640                   // 512 emb + 90 prev + 38 zero pad
#define NOUTPAD  128                   // out_w padded rows

__device__ __forceinline__ float sigf(float x) { return 1.f / (1.f + __expf(-x)); }

// A fragment (16x32 MxK, bf16): lane m holds K at [kA..kA+15]
__device__ __forceinline__ v16bf load_frag_a(const bf16* __restrict__ p) {
  v8bf lo = *(const v8bf*)(p);
  v8bf hi = *(const v8bf*)(p + 8);
  v16bf r;
#pragma unroll
  for (int i = 0; i < 8; ++i) { r[i] = lo[i]; r[i + 8] = hi[i]; }
  return r;
}

// B fragment (32x16 KxN, bf16): lane n holds 16 contiguous K values
__device__ __forceinline__ v16bf load_frag_b(const bf16* __restrict__ p) {
  return *(const v16bf*)(p);
}

// Accumulate a 32x64 wave tile: 2 A frags x 4 B frags -> 8 WMMAs per K chunk
__device__ __forceinline__ void mm_accum(
    v8f acc[8], const bf16* __restrict__ A, int lda,
    const bf16* __restrict__ B, int ldb, int K,
    int m0, int n0, int nl, int kA, int kB)
{
  const bf16* ap0 = A + (size_t)(m0 + nl) * lda + kA;
  const bf16* ap1 = ap0 + (size_t)16 * lda;
  const bf16* bp0 = B + (size_t)(n0 +  0 + nl) * ldb + kB;
  const bf16* bp1 = B + (size_t)(n0 + 16 + nl) * ldb + kB;
  const bf16* bp2 = B + (size_t)(n0 + 32 + nl) * ldb + kB;
  const bf16* bp3 = B + (size_t)(n0 + 48 + nl) * ldb + kB;
  for (int k = 0; k < K; k += 32) {
    v16bf a0 = load_frag_a(ap0 + k);
    v16bf a1 = load_frag_a(ap1 + k);
    v16bf b0 = load_frag_b(bp0 + k);
    acc[0] = __builtin_amdgcn_wmma_f32_16x16x32_bf16(false, a0, false, b0, (short)0, acc[0], false, false);
    acc[4] = __builtin_amdgcn_wmma_f32_16x16x32_bf16(false, a1, false, b0, (short)0, acc[4], false, false);
    v16bf b1 = load_frag_b(bp1 + k);
    acc[1] = __builtin_amdgcn_wmma_f32_16x16x32_bf16(false, a0, false, b1, (short)0, acc[1], false, false);
    acc[5] = __builtin_amdgcn_wmma_f32_16x16x32_bf16(false, a1, false, b1, (short)0, acc[5], false, false);
    v16bf b2 = load_frag_b(bp2 + k);
    acc[2] = __builtin_amdgcn_wmma_f32_16x16x32_bf16(false, a0, false, b2, (short)0, acc[2], false, false);
    acc[6] = __builtin_amdgcn_wmma_f32_16x16x32_bf16(false, a1, false, b2, (short)0, acc[6], false, false);
    v16bf b3 = load_frag_b(bp3 + k);
    acc[3] = __builtin_amdgcn_wmma_f32_16x16x32_bf16(false, a0, false, b3, (short)0, acc[3], false, false);
    acc[7] = __builtin_amdgcn_wmma_f32_16x16x32_bf16(false, a1, false, b3, (short)0, acc[7], false, false);
  }
}

// Generic NT GEMM: C[M,N] = A1[M,K1] * B1[N,K1]^T (+ A2[M,K2] * B2[N,K2]^T) + bias
// 128 threads = 4 waves in 2x2; block tile 64x128; wave tile 32x64.
// mode 0: store f32 (LSTM gates)           -> f32out, ldf32
// mode 1: bias, store bf16                 -> bf16out, ldbf
// mode 2: bias, tanh, store bf16           -> bf16out, ldbf
// mode 3: bias (n<90), store bf16 feedback + sigmoid f32 to output[b, step, n]
// mode 4: bias, tanh, store f32 AND bf16   (s0 initial state)
__global__ __launch_bounds__(128) void wmma_gemm_k(
    const bf16* __restrict__ A1, int lda1, const bf16* __restrict__ B1, int ldb1, int K1,
    const bf16* __restrict__ A2, int lda2, const bf16* __restrict__ B2, int ldb2, int K2,
    const float* __restrict__ bias,
    float* __restrict__ f32out, int ldf32,
    bf16* __restrict__ bf16out, int ldbf,
    float* __restrict__ sigout, int step,
    int mode)
{
  const int lane = threadIdx.x & 31;
  const int wave = threadIdx.x >> 5;
  const int wn = wave & 1;       // N half within block
  const int wm = wave >> 1;      // M half within block
  const int m0 = blockIdx.y * 64 + wm * 32;
  const int n0 = blockIdx.x * 128 + wn * 64;
  const int nl = lane & 15;
  const int hi = lane >> 4;
  const int kA = hi * 16;    // A half-K offset (contiguous K, matches B)
  const int kB = hi * 16;    // B half-K offset (contiguous K, matches A)

  v8f acc[8];
#pragma unroll
  for (int t = 0; t < 8; ++t)
#pragma unroll
    for (int j = 0; j < 8; ++j) acc[t][j] = 0.f;

  mm_accum(acc, A1, lda1, B1, ldb1, K1, m0, n0, nl, kA, kB);
  if (A2) mm_accum(acc, A2, lda2, B2, ldb2, K2, m0, n0, nl, kA, kB);

  // C/D layout: lanes 0-15: N=lane, M=j; lanes 16-31: N=lane-16, M=8+j
#pragma unroll
  for (int mt = 0; mt < 2; ++mt) {
    const int mb = m0 + mt * 16 + hi * 8;
#pragma unroll
    for (int t = 0; t < 4; ++t) {
      const int n = n0 + t * 16 + nl;
      const v8f av = acc[mt * 4 + t];
      if (mode == 3) {
        if (n < OUTD) {
          const float bv = bias[n];
#pragma unroll
          for (int j = 0; j < 8; ++j) {
            const int m = mb + j;
            const float v = av[j] + bv;
            bf16out[(size_t)m * ldbf + n] = (bf16)v;              // prev feedback into x
            sigout[(size_t)m * (SEQ * OUTD) + (size_t)step * OUTD + n] = sigf(v);
          }
        }
      } else {
        const float bv = bias ? bias[n] : 0.f;
#pragma unroll
        for (int j = 0; j < 8; ++j) {
          const int m = mb + j;
          float v = av[j] + bv;
          if (mode == 0) {
            f32out[(size_t)m * ldf32 + n] = v;
          } else if (mode == 1) {
            bf16out[(size_t)m * ldbf + n] = (bf16)v;
          } else if (mode == 2) {
            bf16out[(size_t)m * ldbf + n] = (bf16)tanhf(v);
          } else { // mode 4
            const float tv = tanhf(v);
            f32out[(size_t)m * ldf32 + n] = tv;
            bf16out[(size_t)m * ldbf + n] = (bf16)tv;
          }
        }
      }
    }
  }
}

// LSTM pointwise: gates f32 [B,4H] (i,f,g,o), c f32 in/out, h bf16 out
__global__ __launch_bounds__(256) void lstm_pw_k(
    const float* __restrict__ gates, float* __restrict__ c,
    bf16* __restrict__ h, int H, int total)
{
  const int idx = blockIdx.x * blockDim.x + threadIdx.x;
  if (idx >= total) return;
  const int b = idx / H;
  const int j = idx - b * H;
  const float* g = gates + (size_t)b * 4 * H;
  const float ig = sigf(g[j]);
  const float fg = sigf(g[H + j]);
  const float gg = tanhf(g[2 * H + j]);
  const float og = sigf(g[3 * H + j]);
  const float cn = fg * c[idx] + ig * gg;
  c[idx] = cn;
  h[idx] = (bf16)(og * tanhf(cn));
}

// f32 -> bf16 conversion with optional 2D zero padding (dst [drows,dcols])
__global__ __launch_bounds__(256) void cvt_pad_k(
    const float* __restrict__ src, bf16* __restrict__ dst,
    int srows, int scols, int drows, int dcols)
{
  const int idx = blockIdx.x * blockDim.x + threadIdx.x;
  const int total = drows * dcols;
  if (idx >= total) return;
  const int r = idx / dcols;
  const int cidx = idx - r * dcols;
  const float v = (r < srows && cidx < scols) ? src[(size_t)r * scols + cidx] : 0.f;
  dst[idx] = (bf16)v;
}

__global__ __launch_bounds__(256) void bias_add2_k(
    const float* __restrict__ a, const float* __restrict__ b,
    float* __restrict__ o, int n)
{
  const int idx = blockIdx.x * blockDim.x + threadIdx.x;
  if (idx < n) o[idx] = a[idx] + b[idx];
}

__global__ __launch_bounds__(256) void zero_bf16_k(bf16* __restrict__ p, int n) {
  const int idx = blockIdx.x * blockDim.x + threadIdx.x;
  if (idx < n) p[idx] = (bf16)0.f;
}

// zero bottom-decoder state (h bf16 x2, c f32 x2) in one launch
__global__ __launch_bounds__(256) void zero_bottom_k(
    bf16* __restrict__ h1, bf16* __restrict__ h2,
    float* __restrict__ c1, float* __restrict__ c2, int n)
{
  const int idx = blockIdx.x * blockDim.x + threadIdx.x;
  if (idx < n) {
    h1[idx] = (bf16)0.f; h2[idx] = (bf16)0.f;
    c1[idx] = 0.f;       c2[idx] = 0.f;
  }
}

extern "C" void kernel_launch(void* const* d_in, const int* in_sizes, int n_in,
                              void* d_out, int out_size, void* d_ws, size_t ws_size,
                              hipStream_t stream) {
  (void)in_sizes; (void)n_in; (void)out_size; (void)ws_size;

  const float* z      = (const float*)d_in[0];
  // d_in[1] input_sequence: unused by the reference (pure feedback generation)
  const float* fc_z_w = (const float*)d_in[2];
  const float* fc_z_b = (const float*)d_in[3];
  const float* cWih0  = (const float*)d_in[4];
  const float* cWhh0  = (const float*)d_in[5];
  const float* cbih0  = (const float*)d_in[6];
  const float* cbhh0  = (const float*)d_in[7];
  const float* cWih1  = (const float*)d_in[8];
  const float* cWhh1  = (const float*)d_in[9];
  const float* cbih1  = (const float*)d_in[10];
  const float* cbhh1  = (const float*)d_in[11];
  const float* fc1_w  = (const float*)d_in[12];
  const float* fc1_b  = (const float*)d_in[13];
  const float* cfc_w  = (const float*)d_in[14];
  const float* cfc_b  = (const float*)d_in[15];
  const float* bWih0  = (const float*)d_in[16];
  const float* bWhh0  = (const float*)d_in[17];
  const float* bbih0  = (const float*)d_in[18];
  const float* bbhh0  = (const float*)d_in[19];
  const float* bWih1  = (const float*)d_in[20];
  const float* bWhh1  = (const float*)d_in[21];
  const float* bbih1  = (const float*)d_in[22];
  const float* bbhh1  = (const float*)d_in[23];
  const float* out_w  = (const float*)d_in[24];
  const float* out_b  = (const float*)d_in[25];
  float* out = (float*)d_out;

  // ---- workspace carve-up (256B aligned) ----
  char* ws = (char*)d_ws;
  size_t off = 0;
  auto alloc = [&](size_t bytes) -> void* {
    void* p = ws + off;
    off = (off + bytes + 255) & ~(size_t)255;
    return p;
  };
  bf16* zbf    = (bf16*)alloc((size_t)BATCH * LATENT * 2);
  bf16* wz     = (bf16*)alloc((size_t)CONDH * LATENT * 2);
  bf16* wcih0  = (bf16*)alloc((size_t)4 * CONDH * LATENT * 2);
  bf16* wchh0  = (bf16*)alloc((size_t)4 * CONDH * CONDH * 2);
  bf16* wcih1  = (bf16*)alloc((size_t)4 * CONDH * CONDH * 2);
  bf16* wchh1  = (bf16*)alloc((size_t)4 * CONDH * CONDH * 2);
  bf16* wfc1   = (bf16*)alloc((size_t)CONDOUT * CONDH * 2);
  bf16* wcfc   = (bf16*)alloc((size_t)CONDOUT * CONDOUT * 2);
  bf16* wbih0  = (bf16*)alloc((size_t)4 * BOTH * XDIM * 2);   // padded 602->640
  bf16* wbhh0  = (bf16*)alloc((size_t)4 * BOTH * BOTH * 2);
  bf16* wbih1  = (bf16*)alloc((size_t)4 * BOTH * BOTH * 2);
  bf16* wbhh1  = (bf16*)alloc((size_t)4 * BOTH * BOTH * 2);
  bf16* woutp  = (bf16*)alloc((size_t)NOUTPAD * BOTH * 2);    // padded 90->128 rows
  bf16* condin = (bf16*)alloc((size_t)BATCH * CONDOUT * 2);   // cond_in / fc1 out
  bf16* h1     = (bf16*)alloc((size_t)BATCH * CONDH * 2);
  bf16* h2     = (bf16*)alloc((size_t)BATCH * CONDH * 2);
  bf16* xbuf   = (bf16*)alloc((size_t)BATCH * XDIM * 2);      // [emb | prev | pad]
  bf16* bh1    = (bf16*)alloc((size_t)BATCH * BOTH * 2);
  bf16* bh2    = (bf16*)alloc((size_t)BATCH * BOTH * 2);
  float* c1    = (float*)alloc((size_t)BATCH * CONDH * 4);
  float* c2    = (float*)alloc((size_t)BATCH * CONDH * 4);
  float* bc1   = (float*)alloc((size_t)BATCH * BOTH * 4);
  float* bc2   = (float*)alloc((size_t)BATCH * BOTH * 4);
  float* gates = (float*)alloc((size_t)BATCH * 4 * CONDH * 4);
  float* bcb0  = (float*)alloc((size_t)4 * CONDH * 4);
  float* bcb1  = (float*)alloc((size_t)4 * CONDH * 4);
  float* bbb0  = (float*)alloc((size_t)4 * BOTH * 4);
  float* bbb1  = (float*)alloc((size_t)4 * BOTH * 4);

  auto cvt = [&](const float* s, bf16* d, int sr, int sc, int dr, int dc) {
    const int total = dr * dc;
    cvt_pad_k<<<dim3((total + 255) / 256), dim3(256), 0, stream>>>(s, d, sr, sc, dr, dc);
  };
  auto gemm = [&](const bf16* A1, int lda1, const bf16* B1, int ldb1, int K1,
                  const bf16* A2, int lda2, const bf16* B2, int ldb2, int K2,
                  const float* bias, float* f32o, int ldf, bf16* bfo, int ldb_,
                  float* sigo, int step, int N, int mode) {
    wmma_gemm_k<<<dim3(N / 128, BATCH / 64), dim3(128), 0, stream>>>(
        A1, lda1, B1, ldb1, K1, A2, lda2, B2, ldb2, K2,
        bias, f32o, ldf, bfo, ldb_, sigo, step, mode);
  };
  auto pw = [&](float* cc, bf16* hh) {
    const int total = BATCH * CONDH;
    lstm_pw_k<<<dim3((total + 255) / 256), dim3(256), 0, stream>>>(gates, cc, hh, CONDH, total);
  };

  // ---- one-time conversions / padding / bias fusion ----
  cvt(z,      zbf,   BATCH, LATENT, BATCH, LATENT);
  cvt(fc_z_w, wz,    CONDH, LATENT, CONDH, LATENT);
  cvt(cWih0,  wcih0, 4 * CONDH, LATENT, 4 * CONDH, LATENT);
  cvt(cWhh0,  wchh0, 4 * CONDH, CONDH,  4 * CONDH, CONDH);
  cvt(cWih1,  wcih1, 4 * CONDH, CONDH,  4 * CONDH, CONDH);
  cvt(cWhh1,  wchh1, 4 * CONDH, CONDH,  4 * CONDH, CONDH);
  cvt(fc1_w,  wfc1,  CONDOUT, CONDH,  CONDOUT, CONDH);
  cvt(cfc_w,  wcfc,  CONDOUT, CONDOUT, CONDOUT, CONDOUT);
  cvt(bWih0,  wbih0, 4 * BOTH, CONDOUT + OUTD, 4 * BOTH, XDIM);
  cvt(bWhh0,  wbhh0, 4 * BOTH, BOTH, 4 * BOTH, BOTH);
  cvt(bWih1,  wbih1, 4 * BOTH, BOTH, 4 * BOTH, BOTH);
  cvt(bWhh1,  wbhh1, 4 * BOTH, BOTH, 4 * BOTH, BOTH);
  cvt(out_w,  woutp, OUTD, BOTH, NOUTPAD, BOTH);
  bias_add2_k<<<dim3(16), dim3(256), 0, stream>>>(cbih0, cbhh0, bcb0, 4 * CONDH);
  bias_add2_k<<<dim3(16), dim3(256), 0, stream>>>(cbih1, cbhh1, bcb1, 4 * CONDH);
  bias_add2_k<<<dim3(16), dim3(256), 0, stream>>>(bbih0, bbhh0, bbb0, 4 * BOTH);
  bias_add2_k<<<dim3(16), dim3(256), 0, stream>>>(bbih1, bbhh1, bbb1, 4 * BOTH);
  // cond_in0 = 0, x = 0 (prev=0, pad cols stay 0 forever)
  zero_bf16_k<<<dim3((BATCH * CONDOUT + 255) / 256), dim3(256), 0, stream>>>(condin, BATCH * CONDOUT);
  zero_bf16_k<<<dim3((BATCH * XDIM + 255) / 256), dim3(256), 0, stream>>>(xbuf, BATCH * XDIM);

  // ---- s0 = tanh(z @ fc_z_w^T + b) -> c1 (f32) and h1 (bf16); duplicate to layer 2
  gemm(zbf, LATENT, wz, LATENT, LATENT, nullptr, 0, nullptr, 0, 0,
       fc_z_b, c1, CONDH, h1, CONDH, nullptr, 0, CONDH, /*mode*/4);
  hipMemcpyAsync(c2, c1, (size_t)BATCH * CONDH * 4, hipMemcpyDeviceToDevice, stream);
  hipMemcpyAsync(h2, h1, (size_t)BATCH * CONDH * 2, hipMemcpyDeviceToDevice, stream);

  for (int t = 0; t < NUM_SUB; ++t) {
    // conductor layer 0: gates = cond_in@Wih^T + h1@Whh^T + b
    gemm(condin, CONDOUT, wcih0, LATENT, LATENT, h1, CONDH, wchh0, CONDH, CONDH,
         bcb0, gates, 4 * CONDH, nullptr, 0, nullptr, 0, 4 * CONDH, 0);
    pw(c1, h1);
    // conductor layer 1
    gemm(h1, CONDH, wcih1, CONDH, CONDH, h2, CONDH, wchh1, CONDH, CONDH,
         bcb1, gates, 4 * CONDH, nullptr, 0, nullptr, 0, 4 * CONDH, 0);
    pw(c2, h2);
    // cond_out = h2 @ fc1^T + b  -> next cond_in (bf16)
    gemm(h2, CONDH, wfc1, CONDH, CONDH, nullptr, 0, nullptr, 0, 0,
         fc1_b, nullptr, 0, condin, CONDOUT, nullptr, 0, CONDOUT, 1);
    // emb = tanh(cond_out @ cfc^T + b) -> x[:, 0:512]
    gemm(condin, CONDOUT, wcfc, CONDOUT, CONDOUT, nullptr, 0, nullptr, 0, 0,
         cfc_b, nullptr, 0, xbuf, XDIM, nullptr, 0, CONDOUT, 2);
    // reset bottom decoder state each subsequence
    zero_bottom_k<<<dim3((BATCH * BOTH + 255) / 256), dim3(256), 0, stream>>>(
        bh1, bh2, bc1, bc2, BATCH * BOTH);

    for (int s = 0; s < SUB_LEN; ++s) {
      // bottom layer 0: gates = x@bWih0^T + bh1@bWhh0^T + b  (x = [emb|prev|0])
      gemm(xbuf, XDIM, wbih0, XDIM, XDIM, bh1, BOTH, wbhh0, BOTH, BOTH,
           bbb0, gates, 4 * BOTH, nullptr, 0, nullptr, 0, 4 * BOTH, 0);
      pw(bc1, bh1);
      // bottom layer 1
      gemm(bh1, BOTH, wbih1, BOTH, BOTH, bh2, BOTH, wbhh1, BOTH, BOTH,
           bbb1, gates, 4 * BOTH, nullptr, 0, nullptr, 0, 4 * BOTH, 0);
      pw(bc2, bh2);
      // p = bh2 @ out_w^T + b -> bf16 feedback into x[:,512:602], sigmoid -> output
      gemm(bh2, BOTH, woutp, BOTH, BOTH, nullptr, 0, nullptr, 0, 0,
           out_b, nullptr, 0, xbuf + CONDOUT, XDIM, out, t * SUB_LEN + s, NOUTPAD, 3);
    }
  }
}
